// SynthesisNetwork_28037546508460
// MI455X (gfx1250) — compile-verified, matrix operand order checked by reading, not run on hardware
//
#include <hip/hip_runtime.h>
#include <hip/hip_bf16.h>
#include <math.h>

typedef __attribute__((ext_vector_type(16))) _Float16 v16h;
typedef __attribute__((ext_vector_type(8)))  _Float16 h8;
typedef __attribute__((ext_vector_type(8)))  float    v8f;
typedef __attribute__((ext_vector_type(4)))  float    f4;
typedef _Float16 h16;

#define SQRT2F 1.41421356237309515f

// ---------------------------------------------------------------------------
// Wave-level WMMA GEMM, NB 16x16 f32 tiles per wave, K in chunks of 32 f16.
// MODE: 0 = modulated 1x3 conv (packed f16 weights [3][Co16][Ci32], scale folded)
//       1 = up (transposed) conv (same packed layout)
//       2 = linear (PGA): A = h[16,K] f16, B = W[n,k]*scale (f32)
//       3 = style:        A = wlat[16,512] f32, B = lw[n,k]*scale (f32)
// ---------------------------------------------------------------------------
struct GP {
  int M, N, K;          // real rows, real cols, (padded) depth
  int Ci32, lci, W;     // conv: padded in-channels, log2(Ci32), input width
  int Co16;             // conv: padded out-channels
  float scale;
  const h16*   Wp;      // packed conv weights
  const h16*   Ah;      // linear activations
  const float* Af;      // style latent (f32)
  const float* Bf;      // linear/style f32 weights
  const h16*   Bh;      // conv scaled input Xs [B][Ci32][W]
  const float* dvec;
  const float* noise;
  const float* nsp;
  const float* bias;
  float* outF;
  h16*   outH;
  int outW;
  int flags;            // 1 = leaky0.2 (linear), 8 = fp32 final out
};

__device__ __forceinline__ constexpr int kbase(int e) { return (e < 8) ? e : e + 8; }

template <int MODE, int NB>
__global__ __launch_bounds__(32) void gemm_wmma(GP p) {
  const int lane = threadIdx.x & 31;
  const int r    = lane & 15;
  const int half = lane >> 4;
  const int n0 = blockIdx.x * (16 * NB);
  const int m0 = blockIdx.y * 16;
  const int b  = blockIdx.z;
  const int row = m0 + r;

  int  col[NB];
  bool colOK[NB];
#pragma unroll
  for (int j = 0; j < NB; ++j) { col[j] = n0 + j * 16 + r; colOK[j] = col[j] < p.N; }

  const h16* Bh = (MODE <= 1) ? (p.Bh + (size_t)b * p.Ci32 * p.W) : (const h16*)0;

  // branch-free clamped column for linear/style B (NB == 1 in those modes)
  const int   colc = colOK[0] ? col[0] : (p.N - 1);
  const float bsc  = colOK[0] ? p.scale : 0.f;

  v8f acc[NB];
#pragma unroll
  for (int j = 0; j < NB; ++j) { v8f z = {}; acc[j] = z; }

  const int kfull = p.K & ~31;
  for (int k0 = 0; k0 < kfull; k0 += 32) {
    v16h afrag;
    int kt = 0, ci0 = 0;
    if (MODE <= 1) {
      kt  = k0 >> p.lci;
      ci0 = (k0 & (p.Ci32 - 1)) + (half << 3);
      const h16* ap = p.Wp + ((size_t)(kt * p.Co16 + row)) * p.Ci32 + ci0;
      const h8 lo = *(const h8*)ap;
      const h8 hi = *(const h8*)(ap + 16);
#pragma unroll
      for (int e = 0; e < 8; ++e) { afrag[e] = lo[e]; afrag[8 + e] = hi[e]; }
    } else if (MODE == 2) {
      const size_t arow = (size_t)row * p.K + k0 + (half << 3);
      const h8 lo = *(const h8*)(p.Ah + arow);
      const h8 hi = *(const h8*)(p.Ah + arow + 16);
#pragma unroll
      for (int e = 0; e < 8; ++e) { afrag[e] = lo[e]; afrag[8 + e] = hi[e]; }
    } else {
      const size_t arow = (size_t)row * p.K + k0 + (half << 3);
      const f4 a0 = *(const f4*)(p.Af + arow);
      const f4 a1 = *(const f4*)(p.Af + arow + 4);
      const f4 a2 = *(const f4*)(p.Af + arow + 16);
      const f4 a3 = *(const f4*)(p.Af + arow + 20);
#pragma unroll
      for (int e = 0; e < 4; ++e) {
        afrag[e]      = (h16)a0[e];
        afrag[4 + e]  = (h16)a1[e];
        afrag[8 + e]  = (h16)a2[e];
        afrag[12 + e] = (h16)a3[e];
      }
    }

#pragma unroll
    for (int j = 0; j < NB; ++j) {
      v16h bfrag;
      if (MODE >= 2) {
        const size_t brow = (size_t)colc * p.K + k0 + (half << 3);
        const f4 b0 = *(const f4*)(p.Bf + brow);
        const f4 b1 = *(const f4*)(p.Bf + brow + 4);
        const f4 b2 = *(const f4*)(p.Bf + brow + 16);
        const f4 b3 = *(const f4*)(p.Bf + brow + 20);
#pragma unroll
        for (int e = 0; e < 4; ++e) {
          bfrag[e]      = (h16)(b0[e] * bsc);
          bfrag[4 + e]  = (h16)(b1[e] * bsc);
          bfrag[8 + e]  = (h16)(b2[e] * bsc);
          bfrag[12 + e] = (h16)(b3[e] * bsc);
        }
      } else {
        int x; bool ok;
        if (MODE == 0) {
          x = col[j] + kt - 1;
          ok = colOK[j] && ((unsigned)x < (unsigned)p.W);
        } else {
          const int tp = col[j] - kt;
          x = tp >> 1;
          ok = colOK[j] && (tp >= 0) && !(tp & 1) && (x < p.W);
        }
        if (!ok) {
#pragma unroll
          for (int e = 0; e < 16; ++e) bfrag[e] = (h16)0.f;
        } else {
          const h16* bp = Bh + (size_t)ci0 * p.W + x;
          const size_t sw = (size_t)p.W;
#pragma unroll
          for (int e = 0; e < 16; ++e) bfrag[e] = bp[kbase(e) * sw];
        }
      }
      acc[j] = __builtin_amdgcn_wmma_f32_16x16x32_f16(
          false, afrag, false, bfrag, (short)0, acc[j], false, false);
    }
  }

  // guarded tail chunk (linear K=16/8 only)
  if (MODE == 2 && (p.K & 31)) {
    const int k0 = kfull;
    v16h afrag, bfrag;
#pragma unroll
    for (int e = 0; e < 16; ++e) {
      const int kk = k0 + kbase(e) + (half << 3);
      afrag[e] = (kk < p.K) ? p.Ah[(size_t)row * p.K + kk] : (h16)0.f;
    }
#pragma unroll
    for (int e = 0; e < 16; ++e) {
      const int kk = k0 + kbase(e) + (half << 3);
      float bv = (kk < p.K) ? p.Bf[(size_t)colc * p.K + kk] * bsc : 0.f;
      bfrag[e] = (h16)bv;
    }
    acc[0] = __builtin_amdgcn_wmma_f32_16x16x32_f16(
        false, afrag, false, bfrag, (short)0, acc[0], false, false);
  }

  // -------- epilogue --------
  // D layout: lane owns column col[j]; VGPR rr holds row m0+rr+8*half.
  // All real M are multiples of 8, so each 8-row group is valid as a unit.
  const int  rg   = m0 + (half << 3);                 // row-group base
  const bool rowg = (MODE <= 1) ? (rg < p.M) : true;  // M == 16 for modes 2/3

  // branch-free row-group vectors (clamped base; only read when rowg true)
  f4 dv0 = {}, dv1 = {}, bw0 = {}, bw1 = {};
  if (MODE <= 1) {
    int rb = rg;
    if (rb > p.M - 8) rb = p.M - 8;
    const float* dp = p.dvec + (size_t)b * p.M + rb;
    dv0 = *(const f4*)dp;
    dv1 = *(const f4*)(dp + 4);
    if (MODE == 0) {
      bw0 = *(const f4*)(p.bias + rb);
      bw1 = *(const f4*)(p.bias + rb + 4);
    }
  }
  const float nsv = (MODE == 0) ? p.nsp[0] : 0.f;

#pragma unroll
  for (int j = 0; j < NB; ++j) {
    const int cc = colOK[j] ? col[j] : (p.N - 1);
    float colv = 0.f;
    if (MODE == 0)      colv = nsv * p.noise[cc];
    else if (MODE == 2) colv = p.bias[cc];
    else if (MODE == 3) colv = p.bias[cc] + 1.0f;

    if (rowg && colOK[j]) {
#pragma unroll
      for (int rr = 0; rr < 8; ++rr) {
        const int orow = rg + rr;
        float v = acc[j][rr];
        const float dvr = (rr < 4) ? dv0[rr] : dv1[rr - 4];
        const float bwr = (rr < 4) ? bw0[rr] : bw1[rr - 4];
        if (MODE <= 1) {
          v *= dvr;
          if (MODE == 0) {
            v += colv + bwr;
            v *= SQRT2F;
            v = v > 0.f ? v : 0.2f * v;
          }
          p.outH[((size_t)b * p.M + orow) * p.outW + col[j]] = (h16)v;
        } else if (MODE == 2) {
          v += colv;
          if (p.flags & 1) v = v > 0.f ? v : 0.2f * v;
          if (p.flags & 8) p.outF[(size_t)orow * p.outW + col[j]] = v;
          else             p.outH[(size_t)orow * p.outW + col[j]] = (h16)v;
        } else {
          p.outF[(size_t)orow * p.outW + col[j]] = v + colv;
        }
      }
    }
  }
}

// pack conv weight f32 [Co,Ci,3] (tr=0) or [Ci,Co,3] (tr=1)
// -> f16 [3][Co16][Ci32], scale folded, zero padded
__global__ void pack_w_kernel(const float* __restrict__ w, h16* __restrict__ out,
                              int Co, int Ci, int Co16, int Ci32, float scale, int tr) {
  int i = blockIdx.x * blockDim.x + threadIdx.x;
  int tot = 3 * Co16 * Ci32;
  if (i >= tot) return;
  int kt  = i / (Co16 * Ci32);
  int rem = i - kt * Co16 * Ci32;
  int co  = rem / Ci32;
  int ci  = rem - co * Ci32;
  float v = 0.f;
  if (co < Co && ci < Ci)
    v = (tr ? w[((size_t)ci * Co + co) * 3 + kt]
            : w[((size_t)co * Ci + ci) * 3 + kt]) * scale;
  out[i] = (h16)v;
}

// d[b,co] = rsqrt( sum_ci (sum_k (w*scale)^2) * s^2 + 1e-8 )
__global__ void demod_kernel(const float* __restrict__ w, const float* __restrict__ s,
                             float* __restrict__ d, int Co, int Ci, float s2, int transposed) {
  int i = blockIdx.x * blockDim.x + threadIdx.x;
  if (i >= 16 * Co) return;
  int b = i / Co, co = i - b * Co;
  float acc = 0.f;
  for (int ci = 0; ci < Ci; ++ci) {
    const float* wp = transposed ? (w + ((size_t)ci * Co + co) * 3)
                                 : (w + ((size_t)co * Ci + ci) * 3);
    float w0 = wp[0], w1 = wp[1], w2 = wp[2];
    float ws2 = (w0 * w0 + w1 * w1 + w2 * w2) * s2;
    float sv = s[b * Ci + ci];
    acc += ws2 * sv * sv;
  }
  d[i] = rsqrtf(acc + 1e-8f);
}

// Xs[b,c,x] = X[b,c,x] * s[b,c], channel-padded to Cpad rows (zeros beyond C)
__global__ void scale_in_kernel(const h16* __restrict__ X, const float* __restrict__ s,
                                h16* __restrict__ Xs, int C, int Cpad, int W) {
  int i = blockIdx.x * blockDim.x + threadIdx.x;
  int tot = 16 * Cpad * W;
  if (i >= tot) return;
  int b = i / (Cpad * W);
  int rem = i - b * Cpad * W;
  int c = rem / W;
  int x = rem - c * W;
  float v = 0.f;
  if (c < C) v = (float)X[((size_t)b * C + c) * W + x] * s[b * C + c];
  Xs[i] = (h16)v;
}

__global__ void scale_const_kernel(const float* __restrict__ cst, const float* __restrict__ s,
                                   h16* __restrict__ Xs, int C, int W) {
  int i = blockIdx.x * blockDim.x + threadIdx.x;
  int tot = 16 * C * W;
  if (i >= tot) return;
  int b = i / (C * W);
  int rem = i - b * C * W;
  int c = rem / W;
  Xs[i] = (h16)(cst[rem] * s[b * C + c]);
}

// depthwise [1,3,3,1]/8 blur (pad 1), width Wo=2W+1 -> 2W, then noise/bias/sqrt2/leaky
__global__ void blur_act_kernel(const h16* __restrict__ Y, h16* __restrict__ X,
                                const float* __restrict__ noise, const float* __restrict__ nsp,
                                const float* __restrict__ bias, int C, int Wo) {
  int Wn = Wo - 1;
  int i = blockIdx.x * blockDim.x + threadIdx.x;
  int tot = 16 * C * Wn;
  if (i >= tot) return;
  int b = i / (C * Wn);
  int rem = i - b * C * Wn;
  int c = rem / Wn;
  int v = rem - c * Wn;
  const h16* yrow = Y + ((size_t)b * C + c) * Wo;
  float ym1 = (v >= 1)     ? (float)yrow[v - 1] : 0.f;
  float y0  = (float)yrow[v];
  float y1  = (float)yrow[v + 1];
  float y2  = (v + 2 < Wo) ? (float)yrow[v + 2] : 0.f;
  float z = 0.125f * (ym1 + y2) + 0.375f * (y0 + y1);
  z += nsp[0] * noise[v] + bias[c];
  z *= SQRT2F;
  X[i] = (h16)(z > 0.f ? z : 0.2f * z);
}

// ---------------------------------------------------------------------------
extern "C" void kernel_launch(void* const* d_in, const int* in_sizes, int n_in,
                              void* d_out, int out_size, void* d_ws, size_t ws_size,
                              hipStream_t stream) {
  (void)in_sizes; (void)n_in; (void)out_size; (void)ws_size;
  static const int CH[9]   = {512, 512, 512, 512, 256, 128, 64, 32, 8};
  static const int WIDE[9] = {16, 32, 64, 128, 256, 512, 1024, 2048, 4096};
  static const int PGAD[9] = {32768, 8192, 2048, 512, 128, 32, 16, 8, 1};

  int t = 0;
  const float* wlat = (const float*)d_in[t++];
  const float* cst  = (const float*)d_in[t++];
  const float* c0w  = (const float*)d_in[t++];
  const float* c0lw = (const float*)d_in[t++];
  const float* c0lb = (const float*)d_in[t++];
  const float* n0   = (const float*)d_in[t++];
  const float* ns0  = (const float*)d_in[t++];
  const float* bi0  = (const float*)d_in[t++];
  struct BlkP { const float *uw,*ulw,*ulb,*unoise,*uns,*ubias,*cw,*clw,*clb,*noise,*ns,*bias; } blk[9];
  for (int i = 1; i < 9; ++i) {
    blk[i].uw     = (const float*)d_in[t++];
    blk[i].ulw    = (const float*)d_in[t++];
    blk[i].ulb    = (const float*)d_in[t++];
    blk[i].unoise = (const float*)d_in[t++];
    blk[i].uns    = (const float*)d_in[t++];
    blk[i].ubias  = (const float*)d_in[t++];
    blk[i].cw     = (const float*)d_in[t++];
    blk[i].clw    = (const float*)d_in[t++];
    blk[i].clb    = (const float*)d_in[t++];
    blk[i].noise  = (const float*)d_in[t++];
    blk[i].ns     = (const float*)d_in[t++];
    blk[i].bias   = (const float*)d_in[t++];
  }
  const float *pw[8], *pb[8];
  for (int l = 0; l < 8; ++l) { pw[l] = (const float*)d_in[t++]; pb[l] = (const float*)d_in[t++]; }

  struct CD { int Ci, Co, tr, ci32, co16; const float *w, *lw, *lb; int soff, doff; size_t wpoff; };
  CD cd[17];
  int nc = 0, so = 0, dofs = 0;
  size_t wpe = 0;
  auto addc = [&](int Ci, int Co, int tr, const float* w, const float* lw, const float* lb) {
    CD& c = cd[nc];
    c.Ci = Ci; c.Co = Co; c.tr = tr;
    c.ci32 = (Ci + 31) & ~31; c.co16 = (Co + 15) & ~15;
    c.w = w; c.lw = lw; c.lb = lb;
    c.soff = so; c.doff = dofs; c.wpoff = wpe;
    ++nc; so += 16 * Ci; dofs += 16 * Co;
    wpe += (size_t)3 * c.co16 * c.ci32;
  };
  addc(512, 512, 0, c0w, c0lw, c0lb);
  for (int i = 1; i < 9; ++i) {
    addc(CH[i - 1], CH[i], 1, blk[i].uw, blk[i].ulw, blk[i].ulb);
    addc(CH[i],     CH[i], 0, blk[i].cw, blk[i].clw, blk[i].clb);
  }

  char* base = (char*)d_ws;
  size_t off = 0;
  auto wsalloc = [&](size_t bytes) -> void* {
    void* p = base + off;
    off = (off + bytes + 255) & ~(size_t)255;
    return p;
  };
  float* sbuf  = (float*)wsalloc((size_t)so * sizeof(float));
  float* dbuf  = (float*)wsalloc((size_t)dofs * sizeof(float));
  h16*   wpack = (h16*)wsalloc(wpe * 2);
  const size_t ACT_CAP  = 1060864;   // bufA / bufY
  const size_t PADS_CAP = 2097152;   // bufS: max 16*Ci32*W = 16*32*4096
  h16* bufA = (h16*)wsalloc(ACT_CAP * 2);
  h16* bufS = (h16*)wsalloc(PADS_CAP * 2);
  h16* bufY = (h16*)wsalloc(ACT_CAP * 2);

  auto ilog2 = [](int x) { int l = 0; while ((1 << l) < x) ++l; return l; };

  // 0) pack all conv weights to f16 (scale folded, zero padded)
  for (int j = 0; j < nc; ++j) {
    int tot = 3 * cd[j].co16 * cd[j].ci32;
    pack_w_kernel<<<(tot + 255) / 256, 256, 0, stream>>>(
        cd[j].w, wpack + cd[j].wpoff, cd[j].Co, cd[j].Ci, cd[j].co16, cd[j].ci32,
        1.0f / sqrtf(3.0f * cd[j].Ci), cd[j].tr);
  }
  // 1) style projections
  for (int j = 0; j < nc; ++j) {
    GP p = {};
    p.M = 16; p.N = cd[j].Ci; p.K = 512;
    p.scale = 1.0f / sqrtf(512.0f);
    p.Af = wlat; p.Bf = cd[j].lw; p.bias = cd[j].lb;
    p.outF = sbuf + cd[j].soff; p.outW = cd[j].Ci;
    dim3 g((p.N + 15) / 16, 1, 1);
    gemm_wmma<3, 1><<<g, dim3(32), 0, stream>>>(p);
  }
  // 2) demod
  for (int j = 0; j < nc; ++j) {
    int tot = 16 * cd[j].Co;
    demod_kernel<<<(tot + 255) / 256, 256, 0, stream>>>(
        cd[j].w, sbuf + cd[j].soff, dbuf + cd[j].doff,
        cd[j].Co, cd[j].Ci, 1.0f / (3.0f * cd[j].Ci), cd[j].tr);
  }

  auto conv_gp = [&](CD& c, int N, int W, const h16* Bh, h16* out) {
    GP p = {};
    p.M = c.Co; p.N = N; p.K = 3 * c.ci32;
    p.Ci32 = c.ci32; p.lci = ilog2(c.ci32); p.W = W; p.Co16 = c.co16;
    p.Wp = wpack + c.wpoff; p.Bh = Bh; p.dvec = dbuf + c.doff;
    p.outH = out; p.outW = N;
    return p;
  };
  auto launch_conv = [&](int MODE, GP& p) {
    if (p.N >= 64) {
      dim3 g((p.N + 63) / 64, p.Co16 / 16, 16);
      if (MODE == 0) gemm_wmma<0, 4><<<g, dim3(32), 0, stream>>>(p);
      else           gemm_wmma<1, 4><<<g, dim3(32), 0, stream>>>(p);
    } else {
      dim3 g((p.N + 15) / 16, p.Co16 / 16, 16);
      if (MODE == 0) gemm_wmma<0, 1><<<g, dim3(32), 0, stream>>>(p);
      else           gemm_wmma<1, 1><<<g, dim3(32), 0, stream>>>(p);
    }
  };

  // 3) block 0 (Ci32 == 512, no padding needed for the constant input)
  {
    int C = 512, W = 16, tot = 16 * C * W;
    scale_const_kernel<<<(tot + 255) / 256, 256, 0, stream>>>(cst, sbuf + cd[0].soff, bufS, C, W);
    GP p = conv_gp(cd[0], 16, 16, bufS, bufA);
    p.noise = n0; p.nsp = ns0; p.bias = bi0;
    launch_conv(0, p);
  }

  // 4) blocks 1..8
  int cidx = 1;
  for (int i = 1; i < 9; ++i) {
    int ci = CH[i - 1], co = CH[i];
    int Wi = WIDE[i - 1], Wo = 2 * Wi + 1, Wn = WIDE[i];
    CD& up = cd[cidx++];
    CD& cv = cd[cidx++];
    {
      int tot = 16 * up.ci32 * Wi;
      scale_in_kernel<<<(tot + 255) / 256, 256, 0, stream>>>(
          bufA, sbuf + up.soff, bufS, ci, up.ci32, Wi);
    }
    {
      GP p = conv_gp(up, Wo, Wi, bufS, bufY);
      launch_conv(1, p);
    }
    {
      int tot = 16 * co * Wn;
      blur_act_kernel<<<(tot + 255) / 256, 256, 0, stream>>>(
          bufY, bufA, blk[i].unoise, blk[i].uns, blk[i].ubias, co, Wo);
    }
    {
      int tot = 16 * cv.ci32 * Wn;
      scale_in_kernel<<<(tot + 255) / 256, 256, 0, stream>>>(
          bufA, sbuf + cv.soff, bufS, co, cv.ci32, Wn);
    }
    {
      GP p = conv_gp(cv, Wn, Wn, bufS, bufA);
      p.noise = blk[i].noise; p.nsp = blk[i].ns; p.bias = blk[i].bias;
      launch_conv(0, p);
    }
  }

  // 5) PGA head
  const h16* hin = bufA;
  h16* pA = bufS;
  h16* pB = bufY;
  for (int l = 0; l < 8; ++l) {
    int din = PGAD[l], dout = PGAD[l + 1];
    bool last = (l == 7);
    GP p = {};
    p.M = 16; p.N = dout; p.K = din;
    p.scale = 1.0f / sqrtf((float)din);
    p.Ah = hin; p.Bf = pw[l]; p.bias = pb[l];
    p.outW = dout;
    if (last) { p.flags = 8; p.outF = (float*)d_out; }
    else      { p.flags = 1; p.outH = pA; }
    dim3 g((dout + 15) / 16, 1, 1);
    gemm_wmma<2, 1><<<g, dim3(32), 0, stream>>>(p);
    hin = pA;
    h16* tmp = pA; pA = pB; pB = tmp;
  }
}
